// AlgoMiniBatch_4363686773176
// MI455X (gfx1250) — matrix operand order repaired and checked by reading, not compiled, not run on hardware
//
#include <hip/hip_runtime.h>
#include <hip/hip_bf16.h>

typedef __attribute__((ext_vector_type(16))) _Float16 v16h;
typedef __attribute__((ext_vector_type(8)))  _Float16 v8h;
typedef __attribute__((ext_vector_type(4)))  _Float16 v4h;
typedef __attribute__((ext_vector_type(8)))  float    v8f;

#define B_NODES 4096
#define S1 10
#define L1ROWS (B_NODES * 11)   // 45056
#define FDIM 128
#define KDIM 256

// ---------------------------------------------------------------------------
// Convert W0/W1 (each 128x256 f32) to f16.
// ---------------------------------------------------------------------------
__launch_bounds__(256)
__global__ void cvt_w_kernel(const float* __restrict__ W0, const float* __restrict__ W1,
                             _Float16* __restrict__ w0h, _Float16* __restrict__ w1h) {
    int i = blockIdx.x * 256 + threadIdx.x;   // 0..32767
    w0h[i] = (_Float16)W0[i];
    w1h[i] = (_Float16)W1[i];
}

// ---------------------------------------------------------------------------
// Gather + neighbor mean: builds x0 [45056, 256] f16.
// One wave per row; lane handles 4 feature dims (float4, fully coalesced).
// ---------------------------------------------------------------------------
__launch_bounds__(128)
__global__ void gather_mean_kernel(const float4* __restrict__ feat4,
                                   const int* __restrict__ nodes,
                                   const int* __restrict__ nbr1,
                                   const int* __restrict__ nbr0,
                                   _Float16* __restrict__ x0) {
    const int wave = threadIdx.x >> 5;
    const int lane = threadIdx.x & 31;
    const int row  = blockIdx.x * 4 + wave;       // [0, 45056)
    const int b    = row / 11;
    const int j    = row - b * 11;

    const int self_idx = (j == 0) ? nodes[b] : nbr1[b * S1 + (j - 1)];
    float4 sv = feat4[(size_t)self_idx * 32 + lane];

    const int* nb = nbr0 + (size_t)row * 25;
    float4 acc = make_float4(0.f, 0.f, 0.f, 0.f);
    #pragma unroll 5
    for (int s = 0; s < 25; ++s) {
        float4 v = feat4[(size_t)nb[s] * 32 + lane];
        acc.x += v.x; acc.y += v.y; acc.z += v.z; acc.w += v.w;
    }
    const float inv25 = 0.04f;

    _Float16* o = x0 + (size_t)row * KDIM;
    v4h s16, m16;
    s16[0] = (_Float16)sv.x;  s16[1] = (_Float16)sv.y;
    s16[2] = (_Float16)sv.z;  s16[3] = (_Float16)sv.w;
    m16[0] = (_Float16)(acc.x * inv25); m16[1] = (_Float16)(acc.y * inv25);
    m16[2] = (_Float16)(acc.z * inv25); m16[3] = (_Float16)(acc.w * inv25);
    *(v4h*)(o + lane * 4)       = s16;
    *(v4h*)(o + 128 + lane * 4) = m16;
}

// ---------------------------------------------------------------------------
// out[m,n] = l2norm_row( relu( X[m,:] . W[n,:] + bias[n] ) )
// X: [M,256] f16 row-major, W: [128,256] f16 row-major.
// Block = 256 threads = 8 waves; block owns 16 rows; wave w owns cols 16w..16w+15.
// K=256 walked in 8 steps of 32 with V_WMMA_F32_16X16X32_F16.
//
// CDNA5 wave32 fragment layouts (ISA 7.12.2):
//   A 16x32 f16: lane = (half,hl), m=hl; halves of v16h map to
//                k = k0 + 8*half + [0..7]  and  k = k0 + 16 + 8*half + [0..7]
//   B 32x16 f16: n = hl; k = k0 + 8*half + [0..7]  and  k = k0 + 16 + 8*half + [0..7]
//   C/D f32:     VGPR r -> m = r + 8*half, n = hl
// ---------------------------------------------------------------------------
__launch_bounds__(256)
__global__ void gemm_relu_l2norm_kernel(const _Float16* __restrict__ X,
                                        const _Float16* __restrict__ Wh,
                                        const float* __restrict__ bias,
                                        _Float16* __restrict__ outH,
                                        float* __restrict__ outF,
                                        int storeF32) {
    __shared__ float norm2[16];
    const int wave = threadIdx.x >> 5;   // 0..7
    const int lane = threadIdx.x & 31;
    const int half = lane >> 4;          // 0/1
    const int hl   = lane & 15;
    const int m0   = blockIdx.x * 16;
    const int n0   = wave * 16;

    if (threadIdx.x < 16) norm2[threadIdx.x] = 0.0f;
    __syncthreads();

    const _Float16* xrow = X  + (size_t)(m0 + hl) * KDIM;
    const _Float16* wrow = Wh + (size_t)(n0 + hl) * KDIM;

    v8f c = {};
    #pragma unroll
    for (int kk = 0; kk < 8; ++kk) {
        const int k0 = kk * 32;
        v8h alo = *(const v8h*)(xrow + k0 + 8 * half);
        v8h ahi = *(const v8h*)(xrow + k0 + 16 + 8 * half);
        v8h blo = *(const v8h*)(wrow + k0 + 8 * half);
        v8h bhi = *(const v8h*)(wrow + k0 + 16 + 8 * half);
        v16h a, b;
        #pragma unroll
        for (int i = 0; i < 8; ++i) {
            a[i] = alo[i]; a[8 + i] = ahi[i];
            b[i] = blo[i]; b[8 + i] = bhi[i];
        }
        c = __builtin_amdgcn_wmma_f32_16x16x32_f16(
                /*neg_a=*/false, a, /*neg_b=*/false, b,
                /*c_mod=*/(short)0, c, /*reuse_a=*/false, /*reuse_b=*/false);
    }

    // bias + relu + per-row sum-of-squares (16 cols of this wave's tile)
    const float bn = bias[n0 + hl];
    float ss[8];
    #pragma unroll
    for (int r = 0; r < 8; ++r) {
        float v = fmaxf(c[r] + bn, 0.0f);
        c[r] = v;
        float s = v * v;
        #pragma unroll
        for (int off = 1; off < 16; off <<= 1)
            s += __shfl_xor(s, off, 32);      // stays within each 16-lane half
        ss[r] = s;
    }
    if (hl == 0) {
        #pragma unroll
        for (int r = 0; r < 8; ++r)
            atomicAdd(&norm2[r + 8 * half], ss[r]);
    }
    __syncthreads();

    #pragma unroll
    for (int r = 0; r < 8; ++r) {
        const int m = r + 8 * half;
        const float scale = 1.0f / fmaxf(sqrtf(norm2[m]), 1e-12f);
        const float v = c[r] * scale;
        const size_t idx = (size_t)(m0 + m) * FDIM + n0 + hl;
        if (storeF32) outF[idx] = v;
        else          outH[idx] = (_Float16)v;
    }
}

// ---------------------------------------------------------------------------
// x1[b] = concat( h1[b*11+0, :], mean_{j=1..10} h1[b*11+j, :] )  -> f16 [4096,256]
// ---------------------------------------------------------------------------
__launch_bounds__(128)
__global__ void build_x1_kernel(const _Float16* __restrict__ h1,
                                _Float16* __restrict__ x1) {
    const int b = blockIdx.x;
    const int t = threadIdx.x;               // 0..127
    const _Float16* base = h1 + (size_t)b * 11 * FDIM;
    float self = (float)base[t];
    float acc = 0.0f;
    #pragma unroll
    for (int j = 1; j <= 10; ++j) acc += (float)base[j * FDIM + t];
    x1[(size_t)b * KDIM + t]       = (_Float16)self;
    x1[(size_t)b * KDIM + 128 + t] = (_Float16)(acc * 0.1f);
}

// ---------------------------------------------------------------------------
extern "C" void kernel_launch(void* const* d_in, const int* in_sizes, int n_in,
                              void* d_out, int out_size, void* d_ws, size_t ws_size,
                              hipStream_t stream) {
    const float* features = (const float*)d_in[0];   // [N,128]
    const float* W0       = (const float*)d_in[1];   // [128,256]
    const float* b0       = (const float*)d_in[2];   // [128]
    const float* W1       = (const float*)d_in[3];   // [128,256]
    const float* b1       = (const float*)d_in[4];   // [128]
    const int*   nodes    = (const int*)d_in[5];     // [4096]
    const int*   nbr1     = (const int*)d_in[6];     // [4096,10]
    const int*   nbr0     = (const int*)d_in[7];     // [4096,11,25]

    char* ws = (char*)d_ws;
    _Float16* w0h = (_Float16*)(ws);                                   // 64 KB
    _Float16* w1h = (_Float16*)(ws + (size_t)65536);                   // 64 KB
    _Float16* x0  = (_Float16*)(ws + (size_t)131072);                  // 45056*256*2 = 23068672
    _Float16* h1  = (_Float16*)(ws + (size_t)131072 + 23068672);       // 45056*128*2 = 11534336
    _Float16* x1  = (_Float16*)(ws + (size_t)131072 + 23068672 + 11534336); // 4096*256*2

    cvt_w_kernel<<<128, 256, 0, stream>>>(W0, W1, w0h, w1h);
    gather_mean_kernel<<<L1ROWS / 4, 128, 0, stream>>>(
        (const float4*)features, nodes, nbr1, nbr0, x0);
    gemm_relu_l2norm_kernel<<<L1ROWS / 16, 256, 0, stream>>>(
        x0, w0h, b0, h1, nullptr, 0);
    build_x1_kernel<<<B_NODES, 128, 0, stream>>>(h1, x1);
    gemm_relu_l2norm_kernel<<<B_NODES / 16, 256, 0, stream>>>(
        x1, w1h, b1, nullptr, (float*)d_out, 1);
}
